// Attention_13065290515167
// MI455X (gfx1250) — matrix operand order repaired and checked by reading, not compiled, not run on hardware
//
#include <hip/hip_runtime.h>
#include <math.h>

// ---------- feature probes ----------
#if defined(__has_builtin)
# if __has_builtin(__builtin_amdgcn_global_load_async_to_lds_b128)
#  define HAVE_ASYNC_LDS 1
# endif
# if __has_builtin(__builtin_amdgcn_s_wait_asynccnt)
#  define HAVE_WAIT_ASYNC 1
# endif
#endif

// ---------- types ----------
typedef __attribute__((ext_vector_type(16))) __bf16 v16bf;
typedef __attribute__((ext_vector_type(8)))  float  v8f;
typedef __attribute__((ext_vector_type(4)))  int    v4i_t;

struct __align__(16) U128 { unsigned int x, y, z, w; };

union FragBF {
    v16bf v;
    U128  q[2];
};

static __device__ __forceinline__ unsigned short f2bf(float f) {
    unsigned int u = __float_as_uint(f);
    u += 0x7FFFu + ((u >> 16) & 1u);   // round-to-nearest-even
    return (unsigned short)(u >> 16);
}

static __device__ __forceinline__ v8f wmma_bf16(v16bf a, v16bf b, v8f c) {
    return __builtin_amdgcn_wmma_f32_16x16x32_bf16(false, a, false, b,
                                                   (short)0, c, false, false);
}

// async 16B global -> LDS copy (CDNA5 async path), sync fallback
static __device__ __forceinline__ void async_copy_b128(const unsigned short* g,
                                                       unsigned short* l) {
#if defined(HAVE_ASYNC_LDS)
    __builtin_amdgcn_global_load_async_to_lds_b128(
        (__attribute__((address_space(1))) v4i_t*)g,
        (__attribute__((address_space(3))) v4i_t*)l, 0, 0);
#else
    *(U128*)l = *(const U128*)g;
#endif
}

static __device__ __forceinline__ void wait_async_le1() {
#if defined(HAVE_WAIT_ASYNC)
    __builtin_amdgcn_s_wait_asynccnt(1);
#else
    asm volatile("s_wait_asynccnt 0x1" ::: "memory");
#endif
}
static __device__ __forceinline__ void wait_async_le0() {
#if defined(HAVE_WAIT_ASYNC)
    __builtin_amdgcn_s_wait_asynccnt(0);
#else
    asm volatile("s_wait_asynccnt 0x0" ::: "memory");
#endif
}

#define BATCH   4
#define SDIM    2048
#define EDIM    1024
#define NHEAD   16
#define DDIM    64
#define ROWS    (BATCH * SDIM)          // 8192
#define QKVN    (3 * EDIM)              // 3072

// ---------- kernel 1: fp32 -> bf16 convert (x4 vectorized) ----------
__global__ void convert_x_kernel(const float* __restrict__ x,
                                 unsigned short* __restrict__ xb, int n4) {
    int i = blockIdx.x * 256 + threadIdx.x;
    if (i >= n4) return;
    float4 v = *(const float4*)(x + (size_t)i * 4);
    unsigned short* o = xb + (size_t)i * 4;
    o[0] = f2bf(v.x); o[1] = f2bf(v.y); o[2] = f2bf(v.z); o[3] = f2bf(v.w);
}

// ---------- kernel 2: transpose fp32 [rows][cols] -> bf16 [cols][rows] ----------
__global__ void transpose_bf16_kernel(const float* __restrict__ w,
                                      unsigned short* __restrict__ wt,
                                      int rows, int cols) {
    int idx = blockIdx.x * 256 + threadIdx.x;
    if (idx >= rows * cols) return;
    int j = idx / rows;
    int i = idx - j * rows;
    wt[(size_t)j * rows + i] = f2bf(w[(size_t)i * cols + j]);
}

// ---------- shared GEMM strip: 8 waves x 16 rows, one 64-col B panel ----------
// B panel staged per-block into double-buffered LDS with async copies.
// ldsb: 2 * 64 * 32 bf16 = 8KB.  acc: 4 16x16 f32 tiles per wave.
static __device__ __forceinline__ void gemm_strip_64(
        const unsigned short* __restrict__ arow,   // A + (rbase+lr)*EDIM
        const unsigned short* __restrict__ WT,     // [N][EDIM]
        int cbase, unsigned short* ldsb, int tid, v8f acc[4]) {
    const int lane = tid & 31, lr = lane & 15, half = lane >> 4;
    const int col  = tid >> 2, chunk = tid & 3;    // 64 cols x 4 x 16B

    const unsigned short* gsrc = WT + (size_t)(cbase + col) * EDIM + chunk * 8;
    unsigned short* ldst0 = ldsb + col * 32 + chunk * 8;
    unsigned short* ldst1 = ldst0 + 2048;

    async_copy_b128(gsrc, ldst0);                  // panel 0 in flight

    FragBF a_cur;
    a_cur.q[0] = *(const U128*)(arow + half * 8);
    a_cur.q[1] = *(const U128*)(arow + half * 8 + 16);

    for (int i = 0; i < EDIM / 32; ++i) {
        const int  k0   = i * 32;
        const bool more = (i + 1 < EDIM / 32);
        if (more) async_copy_b128(gsrc + k0 + 32, (i & 1) ? ldst0 : ldst1);
        if (more) wait_async_le1(); else wait_async_le0();
        __syncthreads();                           // panel i visible to all waves

        FragBF a_next;
        if (more) {                                // prefetch next A fragment
            a_next.q[0] = *(const U128*)(arow + k0 + 32 + half * 8);
            a_next.q[1] = *(const U128*)(arow + k0 + 32 + half * 8 + 16);
        }

        const unsigned short* bbase = ldsb + ((i & 1) ? 2048 : 0);
#pragma unroll
        for (int t = 0; t < 4; ++t) {
            const unsigned short* bp = bbase + (t * 16 + lr) * 32 + half * 8;
            FragBF b;
            b.q[0] = *(const U128*)bp;
            b.q[1] = *(const U128*)(bp + 16);
            acc[t] = wmma_bf16(a_cur.v, b.v, acc[t]);
        }
        if (more) a_cur = a_next;
        __syncthreads();                           // done reading before overwrite
    }
}

// ---------- kernel 3: QKV GEMM  X[8192,1024] @ W[1024,3072] + b ----------
__global__ __launch_bounds__(256) void qkv_gemm_kernel(
        const unsigned short* __restrict__ X,
        const unsigned short* __restrict__ WT,     // [3072][1024]
        const float* __restrict__ bias,
        unsigned short* __restrict__ wsq,          // bf16 [B,H,S,D]
        unsigned short* __restrict__ wsk,          // bf16 [B,H,S,D]
        unsigned short* __restrict__ wsvt,         // bf16 [B,H,D,S]
        float* __restrict__ outk,                  // f32  [B,H,D,S]
        float* __restrict__ outv) {                // f32  [B,H,S,D]
    __shared__ __align__(16) unsigned short ldsb[2 * 64 * 32];
    const int tid  = threadIdx.x;
    const int lane = tid & 31, widx = tid >> 5;
    const int lr   = lane & 15, half = lane >> 4;
    const int rg   = blockIdx.x / (QKVN / 64);     // 64 row groups of 128
    const int ng   = blockIdx.x - rg * (QKVN / 64);
    const int rbase = rg * 128 + widx * 16;
    const int cbase = ng * 64;

    const unsigned short* arow = X + (size_t)(rbase + lr) * EDIM;
    v8f zero8 = {0.f,0.f,0.f,0.f,0.f,0.f,0.f,0.f};
    v8f acc[4] = {zero8, zero8, zero8, zero8};

    gemm_strip_64(arow, WT, cbase, ldsb, tid, acc);

#pragma unroll
    for (int t = 0; t < 4; ++t) {
        const int j   = cbase + t * 16 + lr;       // [0,3072)
        const int reg = j >> 10;                   // 0=q 1=k 2=v
        const int jj  = j & 1023;
        const int h   = jj >> 6, d = jj & 63;
        const float bj = bias[j];
#pragma unroll
        for (int r = 0; r < 8; ++r) {
            const int sg = rbase + half * 8 + r;
            const int b  = sg >> 11, s = sg & (SDIM - 1);
            const int bh = b * NHEAD + h;
            const float val = acc[t][r] + bj;
            if (reg == 0) {
                wsq[((size_t)bh * SDIM + s) * DDIM + d] = f2bf(val);
            } else if (reg == 1) {
                outk[((size_t)bh * DDIM + d) * SDIM + s] = val;
                wsk [((size_t)bh * SDIM + s) * DDIM + d] = f2bf(val);
            } else {
                outv[((size_t)bh * SDIM + s) * DDIM + d] = val;
                wsvt[((size_t)bh * DDIM + d) * SDIM + s] = f2bf(val);
            }
        }
    }
}

// ---------- kernel 4: causal flash attention, one wave per (b,h,qtile) ----------
__global__ __launch_bounds__(256) void attn_kernel(
        const unsigned short* __restrict__ qb,     // bf16 [B,H,S,D]
        const unsigned short* __restrict__ kb_,    // bf16 [B,H,S,D]
        const unsigned short* __restrict__ vtb,    // bf16 [B,H,D,S]
        unsigned short* __restrict__ obuf) {       // bf16 [B,S,E]
    __shared__ __align__(16) unsigned short Pst[8][16][32];
    const int tid  = threadIdx.x;
    const int lane = tid & 31, widx = tid >> 5;
    const int lr   = lane & 15, half = lane >> 4;
    const int gw   = blockIdx.x * 8 + widx;
    const int qt   = gw & 127;
    const int bh   = gw >> 7;

    const unsigned short* Q  = qb  + ((size_t)bh * SDIM + qt * 16) * DDIM;
    const unsigned short* K  = kb_ + (size_t)bh * SDIM * DDIM;
    const unsigned short* Vt = vtb + (size_t)bh * DDIM * SDIM;

    FragBF qa0, qa1;
    {
        const unsigned short* qrow = Q + (size_t)lr * DDIM;
        const int b0 = half * 16;
        qa0.q[0] = *(const U128*)(qrow + b0);
        qa0.q[1] = *(const U128*)(qrow + b0 + 8);
        qa1.q[0] = *(const U128*)(qrow + 32 + b0);
        qa1.q[1] = *(const U128*)(qrow + 32 + b0 + 8);
    }

    v8f zero8 = {0.f,0.f,0.f,0.f,0.f,0.f,0.f,0.f};
    v8f o[4] = {zero8, zero8, zero8, zero8};
    float m[8], l[8];
#pragma unroll
    for (int r = 0; r < 8; ++r) { m[r] = -__builtin_inff(); l[r] = 0.f; }

    const int qmax = qt * 16 + 15;
    for (int kblk = 0; kblk <= qmax; kblk += 32) {
        v8f s0 = zero8, s1 = zero8;
        {
            const unsigned short* kp0 = K + (size_t)(kblk + lr) * DDIM + half * 16;
            const unsigned short* kp1 = K + (size_t)(kblk + 16 + lr) * DDIM + half * 16;
            FragBF b0a, b0b, b1a, b1b;
            b0a.q[0] = *(const U128*)kp0;        b0a.q[1] = *(const U128*)(kp0 + 8);
            b0b.q[0] = *(const U128*)(kp0 + 32); b0b.q[1] = *(const U128*)(kp0 + 40);
            b1a.q[0] = *(const U128*)kp1;        b1a.q[1] = *(const U128*)(kp1 + 8);
            b1b.q[0] = *(const U128*)(kp1 + 32); b1b.q[1] = *(const U128*)(kp1 + 40);
            s0 = wmma_bf16(qa0.v, b0a.v, s0);
            s0 = wmma_bf16(qa1.v, b0b.v, s0);
            s1 = wmma_bf16(qa0.v, b1a.v, s1);
            s1 = wmma_bf16(qa1.v, b1b.v, s1);
        }

        float p0[8], p1[8];
#pragma unroll
        for (int r = 0; r < 8; ++r) {
            const int M = r + 8 * half;
            const int qrow = qt * 16 + M;
            float x0 = s0[r] * 0.125f;
            float x1 = s1[r] * 0.125f;
            if (kblk + lr > qrow)      x0 = -10000.0f;
            if (kblk + 16 + lr > qrow) x1 = -10000.0f;
            float bm = fmaxf(x0, x1);
#pragma unroll
            for (int off = 1; off < 16; off <<= 1)
                bm = fmaxf(bm, __shfl_xor(bm, off, 32));
            const float nm   = fmaxf(m[r], bm);
            const float corr = __expf(m[r] - nm);
            p0[r] = __expf(x0 - nm);
            p1[r] = __expf(x1 - nm);
            float rs = p0[r] + p1[r];
#pragma unroll
            for (int off = 1; off < 16; off <<= 1)
                rs += __shfl_xor(rs, off, 32);
            l[r] = l[r] * corr + rs;
            m[r] = nm;
            o[0][r] *= corr; o[1][r] *= corr; o[2][r] *= corr; o[3][r] *= corr;
        }

#pragma unroll
        for (int r = 0; r < 8; ++r) {
            const int M = r + 8 * half;
            Pst[widx][M][lr]      = f2bf(p0[r]);
            Pst[widx][M][lr + 16] = f2bf(p1[r]);
        }
        asm volatile("s_wait_dscnt 0" ::: "memory");
        FragBF pa;
        pa.q[0] = *(const U128*)&Pst[widx][lr][half * 8];
        pa.q[1] = *(const U128*)&Pst[widx][lr][half * 8 + 16];

#pragma unroll
        for (int t = 0; t < 4; ++t) {
            const unsigned short* vp =
                Vt + (size_t)(t * 16 + lr) * SDIM + kblk + half * 8;
            FragBF vb;
            vb.q[0] = *(const U128*)vp;
            vb.q[1] = *(const U128*)(vp + 16);
            o[t] = wmma_bf16(pa.v, vb.v, o[t]);
        }
    }

    const int b = bh >> 4, h = bh & 15;
#pragma unroll
    for (int t = 0; t < 4; ++t) {
#pragma unroll
        for (int r = 0; r < 8; ++r) {
            const int M = r + 8 * half;
            const int srow = qt * 16 + M;
            const int col  = h * DDIM + t * 16 + lr;
            obuf[((size_t)b * SDIM + srow) * EDIM + col] = f2bf(o[t][r] / l[r]);
        }
    }
}

// ---------- kernel 5: projection GEMM  O[8192,1024] @ Wp[1024,1024] + b ----------
__global__ __launch_bounds__(256) void proj_gemm_kernel(
        const unsigned short* __restrict__ O,
        const unsigned short* __restrict__ WT,     // [1024][1024]
        const float* __restrict__ bias,
        float* __restrict__ out) {
    __shared__ __align__(16) unsigned short ldsb[2 * 64 * 32];
    const int tid  = threadIdx.x;
    const int lane = tid & 31, widx = tid >> 5;
    const int lr   = lane & 15, half = lane >> 4;
    const int rg   = blockIdx.x >> 4;              // 64 row groups
    const int ng   = blockIdx.x & 15;              // 16 col groups
    const int rbase = rg * 128 + widx * 16;
    const int cbase = ng * 64;

    const unsigned short* arow = O + (size_t)(rbase + lr) * EDIM;
    v8f zero8 = {0.f,0.f,0.f,0.f,0.f,0.f,0.f,0.f};
    v8f acc[4] = {zero8, zero8, zero8, zero8};

    gemm_strip_64(arow, WT, cbase, ldsb, tid, acc);

#pragma unroll
    for (int t = 0; t < 4; ++t) {
        const int j = cbase + t * 16 + lr;
        const float bj = bias[j];
#pragma unroll
        for (int r = 0; r < 8; ++r) {
            const int sg = rbase + half * 8 + r;
            out[(size_t)sg * EDIM + j] = acc[t][r] + bj;
        }
    }
}

// ---------- host launcher ----------
extern "C" void kernel_launch(void* const* d_in, const int* in_sizes, int n_in,
                              void* d_out, int out_size, void* d_ws, size_t ws_size,
                              hipStream_t stream) {
    const float* hs     = (const float*)d_in[0];
    const float* w_attn = (const float*)d_in[1];
    const float* b_attn = (const float*)d_in[2];
    const float* w_proj = (const float*)d_in[3];
    const float* b_proj = (const float*)d_in[4];

    float* out  = (float*)d_out;
    float* outk = out + (size_t)ROWS * EDIM;
    float* outv = outk + (size_t)ROWS * EDIM;

    char* ws = (char*)d_ws;
    size_t off = 0;
    unsigned short* ws_x  = (unsigned short*)(ws + off); off += (size_t)ROWS * EDIM * 2;
    unsigned short* ws_o  = ws_x;   // aliased: X dead after QKV GEMM
    unsigned short* ws_wa = (unsigned short*)(ws + off); off += (size_t)QKVN * EDIM * 2;
    unsigned short* ws_wp = (unsigned short*)(ws + off); off += (size_t)EDIM * EDIM * 2;
    unsigned short* ws_q  = (unsigned short*)(ws + off); off += (size_t)ROWS * EDIM * 2;
    unsigned short* ws_k  = (unsigned short*)(ws + off); off += (size_t)ROWS * EDIM * 2;
    unsigned short* ws_vt = (unsigned short*)(ws + off); off += (size_t)ROWS * EDIM * 2;

    convert_x_kernel<<<(ROWS * EDIM / 4 + 255) / 256, 256, 0, stream>>>(
        hs, ws_x, ROWS * EDIM / 4);
    transpose_bf16_kernel<<<(EDIM * QKVN + 255) / 256, 256, 0, stream>>>(
        w_attn, ws_wa, EDIM, QKVN);
    transpose_bf16_kernel<<<(EDIM * EDIM + 255) / 256, 256, 0, stream>>>(
        w_proj, ws_wp, EDIM, EDIM);

    // 64 row-groups x 48 col-groups
    qkv_gemm_kernel<<<64 * 48, 256, 0, stream>>>(ws_x, ws_wa, b_attn,
                                                 ws_q, ws_k, ws_vt, outk, outv);
    // 4*16*128 = 8192 waves / 8 per block
    attn_kernel<<<1024, 256, 0, stream>>>(ws_q, ws_k, ws_vt, ws_o);
    // 64 row-groups x 16 col-groups
    proj_gemm_kernel<<<64 * 16, 256, 0, stream>>>(ws_o, ws_wp, b_proj, out);
}
